// EnhancedTransformerPolicy_63144609186410
// MI455X (gfx1250) — matrix operand order repaired and checked by reading, not compiled, not run on hardware
//
#include <hip/hip_runtime.h>

// ---------------- problem constants ----------------
#define B_   64
#define S_   256
#define OBS_ 96
#define ACT_ 29
#define D_   512
#define H_   8
#define L_   8
#define FF_  2048
#define HD_  64
#define M_   (B_ * S_)   // 16384 rows

typedef __attribute__((ext_vector_type(16))) _Float16 v16h;
typedef __attribute__((ext_vector_type(8)))  _Float16 v8h;
typedef __attribute__((ext_vector_type(8)))  float    v8f;

// ---------------- device helpers ----------------
__device__ __forceinline__ float gelu_f(float x) {
    return 0.5f * x * (1.0f + erff(x * 0.70710678118654752f));
}

// A-operand fragment: 16x32 f16, row-major source with row stride ld (halfs).
// Layout: lanes 0-15 row M=lane hold K 0..15; lanes 16-31 row M=lane-16 hold K 16..31 (matches load_frag_b).
__device__ __forceinline__ v16h load_frag_a(const _Float16* base, int ld, int lane) {
    const int row  = lane & 15;
    const int koff = (lane >> 4) << 4;           // 0 or 16
    const _Float16* p = base + (size_t)row * ld + koff;
    v8h lo = *reinterpret_cast<const v8h*>(p);
    v8h hi = *reinterpret_cast<const v8h*>(p + 8);
    v16h r;
#pragma unroll
    for (int i = 0; i < 8; ++i) { r[i] = lo[i]; r[i + 8] = hi[i]; }
    return r;
}

// B-operand fragment: 32x16 f16. Source is "transposed" row-major [N, K] with row stride ld:
// lane (0-15) holds column N=lane, K contiguous 0..15; lanes 16-31 same N, K 16..31.
__device__ __forceinline__ v16h load_frag_b(const _Float16* base, int ld, int lane) {
    const int col  = lane & 15;
    const int koff = (lane >> 4) << 4;           // 0 or 16
    return *reinterpret_cast<const v16h*>(base + (size_t)col * ld + koff);
}

__device__ __forceinline__ v8f wmma16(v16h a, v16h b, v8f c) {
    return __builtin_amdgcn_wmma_f32_16x16x32_f16(false, a, false, b, (short)0, c, false, false);
}

// CDNA5 async global->LDS copy (16 bytes per lane), tracked by ASYNCcnt.
__device__ __forceinline__ void async_copy_b128(void* lds_ptr, const void* gptr) {
    unsigned int       l = (unsigned int)(unsigned long long)lds_ptr;   // low 32 bits = LDS offset
    unsigned long long g = (unsigned long long)gptr;
    asm volatile("global_load_async_to_lds_b128 %0, %1, off" :: "v"(l), "v"(g) : "memory");
}

// Stage one 128x64-half A tile and one 128x64-half B tile into LDS (16 async b128 per thread).
__device__ __forceinline__ void stage64(_Float16* Abuf, _Float16* Bbuf,
                                        const _Float16* Ag, const _Float16* Bg,
                                        int K, int k0, int tid) {
#pragma unroll
    for (int j = 0; j < 8; ++j) {
        const int chunk = tid + 128 * j;      // 1024 chunks of 8 halfs per tile
        const int row  = chunk >> 3;
        const int coff = (chunk & 7) * 8;
        async_copy_b128(Abuf + row * 64 + coff, Ag + (size_t)row * K + k0 + coff);
        async_copy_b128(Bbuf + row * 64 + coff, Bg + (size_t)row * K + k0 + coff);
    }
}

// ---------------- weight convert + transpose: dst[n*K+k] = (f16)src[k*N+n] ----------------
__global__ void convt_kernel(const float* __restrict__ src, _Float16* __restrict__ dst,
                             int K, int N) {
    size_t i = (size_t)blockIdx.x * 256 + threadIdx.x;
    if (i < (size_t)K * N) {
        int k = (int)(i / N);
        int n = (int)(i % N);
        dst[(size_t)n * K + k] = (_Float16)src[i];
    }
}

// v [B*S, D] f16 -> vt [B, H, HD, S] f16
__global__ void vtrans_kernel(const _Float16* __restrict__ vh, _Float16* __restrict__ vt) {
    size_t i = (size_t)blockIdx.x * 256 + threadIdx.x;   // over B*S*D
    int col = (int)(i & (D_ - 1));
    size_t row = i >> 9;
    int b = (int)(row >> 8);
    int s = (int)(row & (S_ - 1));
    int h = col >> 6;
    int d = col & (HD_ - 1);
    vt[(((size_t)(b * H_ + h)) * HD_ + d) * S_ + s] = vh[i];
}

// ---------------- WMMA GEMM: out = A[M,K] @ Wt[N,K]^T + bias ----------------
// 4 waves / 128x128 macro-tile; double-buffered LDS fed by async global->LDS copies.
// MODE 0: ->f16   MODE 1: gelu ->f16   MODE 2: +res ->f32   MODE 3: ->f32
template <int MODE>
__global__ __launch_bounds__(128) void gemm_wmma(
    const _Float16* __restrict__ A, const _Float16* __restrict__ Wt,
    const float* __restrict__ bias, const float* __restrict__ res,
    _Float16* __restrict__ outh, float* __restrict__ outf,
    int M, int N, int K)
{
    __shared__ alignas(16) _Float16 At[2][128][64];   // 32 KB
    __shared__ alignas(16) _Float16 Bt[2][128][64];   // 32 KB
    const int tid  = threadIdx.x;
    const int lane = tid & 31;
    const int wave = tid >> 5;
    const int wm = wave >> 1;                 // 0..1
    const int wn = wave & 1;                  // 0..1
    const int nb0 = blockIdx.x * 128;
    const int mb0 = blockIdx.y * 128;
    const _Float16* Ag = A  + (size_t)mb0 * K;
    const _Float16* Bg = Wt + (size_t)nb0 * K;
    const int nsteps = K >> 6;                // K multiple of 64

    const v8f vzero = {};
    v8f acc[4][4];
#pragma unroll
    for (int i = 0; i < 4; ++i)
#pragma unroll
        for (int j = 0; j < 4; ++j) acc[i][j] = vzero;

    stage64(&At[0][0][0], &Bt[0][0][0], Ag, Bg, K, 0, tid);

    for (int step = 0; step < nsteps; ++step) {
        const int buf = step & 1;
        if (step + 1 < nsteps) {
            stage64(&At[buf ^ 1][0][0], &Bt[buf ^ 1][0][0], Ag, Bg, K, (step + 1) << 6, tid);
            asm volatile("s_wait_asynccnt 0x10" ::: "memory");   // current tile's 16 copies done
        } else {
            asm volatile("s_wait_asynccnt 0x0" ::: "memory");
        }
        __syncthreads();                                          // all waves' copies visible
#pragma unroll
        for (int ks = 0; ks < 64; ks += 32) {
            v16h a[4], b[4];
#pragma unroll
            for (int i = 0; i < 4; ++i)
                a[i] = load_frag_a(&At[buf][wm * 64 + 16 * i][ks], 64, lane);
#pragma unroll
            for (int j = 0; j < 4; ++j)
                b[j] = load_frag_b(&Bt[buf][wn * 64 + 16 * j][ks], 64, lane);
#pragma unroll
            for (int i = 0; i < 4; ++i)
#pragma unroll
                for (int j = 0; j < 4; ++j)
                    acc[i][j] = wmma16(a[i], b[j], acc[i][j]);
        }
        __syncthreads();                                          // done reading buf before overwrite
    }

    const int m0 = mb0 + wm * 64;
    const int n0 = nb0 + wn * 64;
    const int rbase = 8 * (lane >> 4);
    const int cl = lane & 15;
#pragma unroll
    for (int i = 0; i < 4; ++i) {
#pragma unroll
        for (int j = 0; j < 4; ++j) {
            const int ncol = n0 + 16 * j + cl;
            const float bv = bias[ncol];
#pragma unroll
            for (int v = 0; v < 8; ++v) {
                const int mrow = m0 + 16 * i + rbase + v;
                const size_t idx = (size_t)mrow * N + ncol;
                float x = acc[i][j][v] + bv;
                if (MODE == 1) x = gelu_f(x);
                if (MODE == 0 || MODE == 1) outh[idx] = (_Float16)x;
                else if (MODE == 2)         outf[idx] = x + res[idx];
                else                        outf[idx] = x;
            }
        }
    }
}

// ---------------- attention: one wave per (b, h, 16-row q tile), S=256, HD=64 ----------------
__global__ __launch_bounds__(32) void attention_kernel(
    const _Float16* __restrict__ qh, const _Float16* __restrict__ kh,
    const _Float16* __restrict__ vt, _Float16* __restrict__ ah)
{
    const int lane = threadIdx.x;
    int id = blockIdx.x;
    const int qt = id & 15; id >>= 4;
    const int h  = id & 7;
    const int b  = id >> 3;

    const _Float16* Q  = qh + ((size_t)b * S_) * D_ + h * HD_;
    const _Float16* Kp = kh + ((size_t)b * S_) * D_ + h * HD_;
    const _Float16* Vt = vt + ((size_t)(b * H_ + h) * HD_) * S_;

    __shared__ alignas(16) _Float16 Pl[16 * S_];

    v16h qa[2];
    qa[0] = load_frag_a(Q + (size_t)(qt * 16) * D_ + 0,  D_, lane);
    qa[1] = load_frag_a(Q + (size_t)(qt * 16) * D_ + 32, D_, lane);

    const v8f vzero = {};
    v8f s[16];
#pragma unroll
    for (int jt = 0; jt < 16; ++jt) {
        v8f a = vzero;
        v16h b0 = load_frag_b(Kp + (size_t)(jt * 16) * D_ + 0,  D_, lane);
        v16h b1 = load_frag_b(Kp + (size_t)(jt * 16) * D_ + 32, D_, lane);
        a = wmma16(qa[0], b0, a);
        a = wmma16(qa[1], b1, a);
        s[jt] = a;
    }

    const float scale = 0.125f;  // 1/sqrt(64)
    float rsum[8];
#pragma unroll
    for (int v = 0; v < 8; ++v) {
        float mx = s[0][v];
#pragma unroll
        for (int jt = 1; jt < 16; ++jt) mx = fmaxf(mx, s[jt][v]);
        mx = fmaxf(mx, __shfl_xor(mx, 1, 32));
        mx = fmaxf(mx, __shfl_xor(mx, 2, 32));
        mx = fmaxf(mx, __shfl_xor(mx, 4, 32));
        mx = fmaxf(mx, __shfl_xor(mx, 8, 32));
        float sm = 0.f;
#pragma unroll
        for (int jt = 0; jt < 16; ++jt) {
            float p = __expf(scale * (s[jt][v] - mx));
            s[jt][v] = p;
            sm += p;
        }
        sm += __shfl_xor(sm, 1, 32);
        sm += __shfl_xor(sm, 2, 32);
        sm += __shfl_xor(sm, 4, 32);
        sm += __shfl_xor(sm, 8, 32);
        rsum[v] = sm;
    }

    // stage P (f16, row-major 16x256) in LDS to re-fragment as A operand
    const int rbase = 8 * (lane >> 4);
    const int cl = lane & 15;
#pragma unroll
    for (int jt = 0; jt < 16; ++jt)
#pragma unroll
        for (int v = 0; v < 8; ++v)
            Pl[(rbase + v) * S_ + jt * 16 + cl] = (_Float16)s[jt][v];
    __syncthreads();

    v8f o[4];
#pragma unroll
    for (int nt = 0; nt < 4; ++nt) o[nt] = vzero;
#pragma unroll
    for (int kc = 0; kc < 8; ++kc) {
        v16h pa = load_frag_a(&Pl[0] + kc * 32, S_, lane);
#pragma unroll
        for (int nt = 0; nt < 4; ++nt) {
            v16h vb = load_frag_b(Vt + (size_t)(nt * 16) * S_ + kc * 32, S_, lane);
            o[nt] = wmma16(pa, vb, o[nt]);
        }
    }

    _Float16* Ob = ah + ((size_t)b * S_ + qt * 16) * D_ + h * HD_;
#pragma unroll
    for (int nt = 0; nt < 4; ++nt)
#pragma unroll
        for (int v = 0; v < 8; ++v)
            Ob[(size_t)(rbase + v) * D_ + nt * 16 + cl] = (_Float16)(o[nt][v] / rsum[v]);
}

// ---------------- LayerNorm (+ optional GELU), one 128-thread block per row ----------------
__global__ __launch_bounds__(128) void ln_kernel(
    const float* __restrict__ in, const float* __restrict__ g, const float* __restrict__ be,
    float* __restrict__ outf, _Float16* __restrict__ outh, int ncols, int do_gelu)
{
    const int row = blockIdx.x;
    const int tid = threadIdx.x;
    const float* p = in + (size_t)row * ncols;
    __shared__ float red[8];

    float sum = 0.f, sq = 0.f;
    for (int c = tid; c < ncols; c += 128) { float x = p[c]; sum += x; sq += x * x; }
#pragma unroll
    for (int off = 1; off < 32; off <<= 1) {
        sum += __shfl_xor(sum, off, 32);
        sq  += __shfl_xor(sq,  off, 32);
    }
    if ((tid & 31) == 0) { red[tid >> 5] = sum; red[4 + (tid >> 5)] = sq; }
    __syncthreads();
    sum = red[0] + red[1] + red[2] + red[3];
    sq  = red[4] + red[5] + red[6] + red[7];

    const float mean = sum / ncols;
    const float var  = sq / ncols - mean * mean;
    const float rstd = rsqrtf(var + 1e-5f);
    for (int c = tid; c < ncols; c += 128) {
        float x = (p[c] - mean) * rstd * g[c] + be[c];
        if (do_gelu) x = gelu_f(x);
        if (outf) outf[(size_t)row * ncols + c] = x;
        outh[(size_t)row * ncols + c] = (_Float16)x;
    }
}

// ---------------- input projection: gelu(ln(obs@W_in+b)) + pos_encoding ----------------
__global__ __launch_bounds__(128) void input_proj_kernel(
    const float* __restrict__ obs, const float* __restrict__ W_in, const float* __restrict__ b_in,
    const float* __restrict__ g_in, const float* __restrict__ be_in,
    float* __restrict__ x32, _Float16* __restrict__ xh)
{
    const int row = blockIdx.x;            // 0..16383
    const int s   = row & (S_ - 1);
    const int tid = threadIdx.x;
    __shared__ float ob[OBS_];
    __shared__ float y[D_];
    __shared__ float red[8];

    const float* o = obs + (size_t)row * OBS_;
    for (int i = tid; i < OBS_; i += 128) ob[i] = o[i];
    __syncthreads();

    for (int c = tid; c < D_; c += 128) {
        float acc = b_in[c];
        for (int k = 0; k < OBS_; ++k) acc += ob[k] * W_in[k * D_ + c];
        y[c] = acc;
    }
    __syncthreads();

    float sum = 0.f, sq = 0.f;
    for (int c = tid; c < D_; c += 128) { float x = y[c]; sum += x; sq += x * x; }
#pragma unroll
    for (int off = 1; off < 32; off <<= 1) {
        sum += __shfl_xor(sum, off, 32);
        sq  += __shfl_xor(sq,  off, 32);
    }
    if ((tid & 31) == 0) { red[tid >> 5] = sum; red[4 + (tid >> 5)] = sq; }
    __syncthreads();
    sum = red[0] + red[1] + red[2] + red[3];
    sq  = red[4] + red[5] + red[6] + red[7];

    const float mean = sum / D_;
    const float rstd = rsqrtf(sq / D_ - mean * mean + 1e-5f);
    for (int c = tid; c < D_; c += 128) {
        float t = gelu_f((y[c] - mean) * rstd * g_in[c] + be_in[c]);
        const int i2 = c >> 1;
        const float div = __expf((float)(2 * i2) * (-9.210340371976184f / (float)D_));
        const float ang = (float)s * div;
        t += (c & 1) ? __cosf(ang) : __sinf(ang);
        x32[(size_t)row * D_ + c] = t;
        xh [(size_t)row * D_ + c] = (_Float16)t;
    }
}

// ---------------- final head: tanh(y@Wp3+b)*scale+bias, N=29 ----------------
__global__ __launch_bounds__(32) void head_final_kernel(
    const _Float16* __restrict__ y2h, const float* __restrict__ Wp3, const float* __restrict__ bp3,
    const float* __restrict__ ascale, const float* __restrict__ abias, float* __restrict__ out)
{
    const int row = blockIdx.x;
    const int c = threadIdx.x;
    if (c < ACT_) {
        float acc = bp3[c];
        const _Float16* yp = y2h + (size_t)row * (D_ / 4);
        for (int k = 0; k < D_ / 4; ++k) acc += (float)yp[k] * Wp3[k * ACT_ + c];
        out[(size_t)row * ACT_ + c] = tanhf(acc) * ascale[c] + abias[c];
    }
}

// ---------------- host orchestration ----------------
extern "C" void kernel_launch(void* const* d_in, const int* in_sizes, int n_in,
                              void* d_out, int out_size, void* d_ws, size_t ws_size,
                              hipStream_t stream) {
    (void)in_sizes; (void)n_in; (void)out_size; (void)ws_size;
    const float* obs   = (const float*)d_in[0];
    const float* W_in  = (const float*)d_in[1];
    const float* b_in  = (const float*)d_in[2];
    const float* g_in  = (const float*)d_in[3];
    const float* be_in = (const float*)d_in[4];
    const float* Wq  = (const float*)d_in[5];
    const float* bq  = (const float*)d_in[6];
    const float* Wk  = (const float*)d_in[7];
    const float* bk  = (const float*)d_in[8];
    const float* Wv  = (const float*)d_in[9];
    const float* bv  = (const float*)d_in[10];
    const float* Wo  = (const float*)d_in[11];
    const float* bo  = (const float*)d_in[12];
    const float* g1  = (const float*)d_in[13];
    const float* be1 = (const float*)d_in[14];
    const float* W1  = (const float*)d_in[15];
    const float* b1  = (const float*)d_in[16];
    const float* W2  = (const float*)d_in[17];
    const float* b2  = (const float*)d_in[18];
    const float* g2  = (const float*)d_in[19];
    const float* be2 = (const float*)d_in[20];
    const float* Wp1 = (const float*)d_in[21];
    const float* bp1 = (const float*)d_in[22];
    const float* gp1 = (const float*)d_in[23];
    const float* bep1= (const float*)d_in[24];
    const float* Wp2 = (const float*)d_in[25];
    const float* bp2 = (const float*)d_in[26];
    const float* gp2 = (const float*)d_in[27];
    const float* bep2= (const float*)d_in[28];
    const float* Wp3 = (const float*)d_in[29];
    const float* bp3 = (const float*)d_in[30];
    const float* ascale = (const float*)d_in[31];
    const float* abias  = (const float*)d_in[32];
    float* out = (float*)d_out;

    char* ws = (char*)d_ws;
    size_t off = 0;
    auto alloc = [&](size_t bytes) -> char* {
        char* p = ws + off;
        off += (bytes + 255) & ~(size_t)255;
        return p;
    };

    const size_t WOFF  = (size_t)D_ * D_;    // 262144
    const size_t W1OFF = (size_t)D_ * FF_;   // 1048576

    _Float16* wq_t  = (_Float16*)alloc((size_t)L_ * WOFF * 2);
    _Float16* wk_t  = (_Float16*)alloc((size_t)L_ * WOFF * 2);
    _Float16* wv_t  = (_Float16*)alloc((size_t)L_ * WOFF * 2);
    _Float16* wo_t  = (_Float16*)alloc((size_t)L_ * WOFF * 2);
    _Float16* w1_t  = (_Float16*)alloc((size_t)L_ * W1OFF * 2);
    _Float16* w2_t  = (_Float16*)alloc((size_t)L_ * W1OFF * 2);
    _Float16* wp1_t = (_Float16*)alloc((size_t)D_ * (D_ / 2) * 2);
    _Float16* wp2_t = (_Float16*)alloc((size_t)(D_ / 2) * (D_ / 4) * 2);
    float*    x32   = (float*)   alloc((size_t)M_ * D_ * 4);
    float*    y32   = (float*)   alloc((size_t)M_ * D_ * 4);
    _Float16* xh    = (_Float16*)alloc((size_t)M_ * D_ * 2);
    // qkv block (64MB); FFN hidden aliases it (qkv dead during FFN)
    char*     qkv   = alloc((size_t)M_ * D_ * 2 * 4);
    _Float16* qh    = (_Float16*)qkv;
    _Float16* kh    = (_Float16*)(qkv + (size_t)M_ * D_ * 2);
    _Float16* vh    = (_Float16*)(qkv + (size_t)M_ * D_ * 4);
    _Float16* vt    = (_Float16*)(qkv + (size_t)M_ * D_ * 6);
    _Float16* hh    = (_Float16*)qkv;                       // [M, FF] alias
    _Float16* ah    = (_Float16*)alloc((size_t)M_ * D_ * 2);
    _Float16* h1h   = (_Float16*)alloc((size_t)M_ * (D_ / 2) * 2);
    _Float16* h2h   = (_Float16*)alloc((size_t)M_ * (D_ / 4) * 2);

    // ---- weight conversion (fp32 -> f16, transposed [N,K]) ----
    const int cgrid_dd = (int)((WOFF + 255) / 256);
    const int cgrid_df = (int)((W1OFF + 255) / 256);
    for (int l = 0; l < L_; ++l) {
        convt_kernel<<<cgrid_dd, 256, 0, stream>>>(Wq + l * WOFF, wq_t + l * WOFF, D_, D_);
        convt_kernel<<<cgrid_dd, 256, 0, stream>>>(Wk + l * WOFF, wk_t + l * WOFF, D_, D_);
        convt_kernel<<<cgrid_dd, 256, 0, stream>>>(Wv + l * WOFF, wv_t + l * WOFF, D_, D_);
        convt_kernel<<<cgrid_dd, 256, 0, stream>>>(Wo + l * WOFF, wo_t + l * WOFF, D_, D_);
        convt_kernel<<<cgrid_df, 256, 0, stream>>>(W1 + l * W1OFF, w1_t + l * W1OFF, D_, FF_);
        convt_kernel<<<cgrid_df, 256, 0, stream>>>(W2 + l * W1OFF, w2_t + l * W1OFF, FF_, D_);
    }
    convt_kernel<<<(D_ * (D_ / 2) + 255) / 256, 256, 0, stream>>>(Wp1, wp1_t, D_, D_ / 2);
    convt_kernel<<<((D_ / 2) * (D_ / 4) + 255) / 256, 256, 0, stream>>>(Wp2, wp2_t, D_ / 2, D_ / 4);

    // ---- input projection + LN + GELU + positional encoding ----
    input_proj_kernel<<<M_, 128, 0, stream>>>(obs, W_in, b_in, g_in, be_in, x32, xh);

    // ---- transformer layers ----
    const dim3 gD(D_ / 128, M_ / 128);    // N=512
    const dim3 gF(FF_ / 128, M_ / 128);   // N=2048
    for (int l = 0; l < L_; ++l) {
        gemm_wmma<0><<<gD, 128, 0, stream>>>(xh, wq_t + l * WOFF, bq + l * D_, nullptr, qh, nullptr, M_, D_, D_);
        gemm_wmma<0><<<gD, 128, 0, stream>>>(xh, wk_t + l * WOFF, bk + l * D_, nullptr, kh, nullptr, M_, D_, D_);
        gemm_wmma<0><<<gD, 128, 0, stream>>>(xh, wv_t + l * WOFF, bv + l * D_, nullptr, vh, nullptr, M_, D_, D_);
        vtrans_kernel<<<(int)(((size_t)M_ * D_) / 256), 256, 0, stream>>>(vh, vt);
        attention_kernel<<<B_ * H_ * (S_ / 16), 32, 0, stream>>>(qh, kh, vt, ah);
        gemm_wmma<2><<<gD, 128, 0, stream>>>(ah, wo_t + l * WOFF, bo + l * D_, x32, nullptr, y32, M_, D_, D_);
        ln_kernel<<<M_, 128, 0, stream>>>(y32, g1 + l * D_, be1 + l * D_, x32, xh, D_, 0);
        gemm_wmma<1><<<gF, 128, 0, stream>>>(xh, w1_t + l * W1OFF, b1 + l * FF_, nullptr, hh, nullptr, M_, FF_, D_);
        gemm_wmma<2><<<gD, 128, 0, stream>>>(hh, w2_t + l * W1OFF, b2 + l * D_, x32, nullptr, y32, M_, D_, FF_);
        ln_kernel<<<M_, 128, 0, stream>>>(y32, g2 + l * D_, be2 + l * D_, x32, xh, D_, 0);
    }

    // ---- policy head ----
    gemm_wmma<3><<<dim3((D_ / 2) / 128, M_ / 128), 128, 0, stream>>>(xh, wp1_t, bp1, nullptr, nullptr, y32, M_, D_ / 2, D_);
    ln_kernel<<<M_, 128, 0, stream>>>(y32, gp1, bep1, nullptr, h1h, D_ / 2, 1);
    gemm_wmma<3><<<dim3((D_ / 4) / 128, M_ / 128), 128, 0, stream>>>(h1h, wp2_t, bp2, nullptr, nullptr, y32, M_, D_ / 4, D_ / 2);
    ln_kernel<<<M_, 128, 0, stream>>>(y32, gp2, bep2, nullptr, h2h, D_ / 4, 1);
    head_final_kernel<<<M_, 32, 0, stream>>>(h2h, Wp3, bp3, ascale, abias, out);
}
